// SparseSelfAttention_33964601376967
// MI455X (gfx1250) — matrix operand order repaired and checked by reading, not compiled, not run on hardware
//
#include <hip/hip_runtime.h>

// CDNA5 / gfx1250, wave32. f16 WMMA GEMM pipeline for the "sparse self attention"
// reference (per-position q.k dots + seq-wide softmax, not full attention).
// Each GEMM wave owns TWO 16-row M-subtiles so every B fragment feeds 2 WMMAs
// (halves L2 B-traffic; the GEMMs are WMMA-pipe bound otherwise).

typedef __attribute__((ext_vector_type(16))) _Float16 v16h;
typedef __attribute__((ext_vector_type(8)))  _Float16 v8h;
typedef __attribute__((ext_vector_type(8)))  float    v8f;

#define DIM       1024
#define HEADS     16
#define DIM_HEAD  64
#define SEQ       4096
#define BATCH     4
#define ROWS      (BATCH * SEQ)   // 16384

// ---------------------------------------------------------------- prep kernels

__global__ void cvt_f32_to_f16(const float* __restrict__ src,
                               _Float16* __restrict__ dst, int n) {
  int i = blockIdx.x * blockDim.x + threadIdx.x;
  int stride = gridDim.x * blockDim.x;
  for (; i < n; i += stride) dst[i] = (_Float16)src[i];
}

// Wt[e][d] = W[e][d]  (f16 copy; B[k,n] = W[n,k], contiguous in k per column n)
__global__ void transpose_to_f16(const float* __restrict__ W,
                                 _Float16* __restrict__ Wt) {
  int i = blockIdx.x * blockDim.x + threadIdx.x;
  int stride = gridDim.x * blockDim.x;
  for (; i < DIM * DIM; i += stride) {
    int d = i >> 10, e = i & (DIM - 1);
    Wt[i] = (_Float16)W[i];
  }
}

// ------------------------------------------------- fused QKV + per-position dots
// One wave = 32 rows (2 M-subtiles) x one head (64 cols). 4 waves / block.
// 24 v_wmma_f32_16x16x32_f16 per K-iteration; q.k reduced in-register.

__global__ __launch_bounds__(128)
void qkv_dots_kernel(const _Float16* __restrict__ xh,
                     const _Float16* __restrict__ Wqt,
                     const _Float16* __restrict__ Wkt,
                     const _Float16* __restrict__ Wvt,
                     _Float16* __restrict__ vh,
                     float* __restrict__ dots) {
  const int lane  = threadIdx.x & 31;
  const int wave  = threadIdx.x >> 5;
  const int mtile = (blockIdx.x * 4 + wave) * 32;
  const int head  = blockIdx.y;
  const int nbase = head * DIM_HEAD;
  const int mrow  = mtile + (lane & 15);
  const int ahalf = (lane < 16) ? 0 : 8;

  v8f zero = {};
  v8f cq[2][4], ck[2][4], cv[2][4];
#pragma unroll
  for (int s = 0; s < 2; ++s)
#pragma unroll
    for (int j = 0; j < 4; ++j) { cq[s][j] = zero; ck[s][j] = zero; cv[s][j] = zero; }

  const _Float16* arow0 = xh + (size_t)mrow * DIM + ahalf;            // subtile 0
  const _Float16* arow1 = arow0 + (size_t)16 * DIM;                   // subtile 1

  for (int kb = 0; kb < DIM; kb += 32) {
    // A fragments (16x32 f16): lane-half selects K sub-range, two 16B loads each
    v8h p0 = *reinterpret_cast<const v8h*>(arow0 + kb);
    v8h p1 = *reinterpret_cast<const v8h*>(arow0 + kb + 16);
    v16h a0 = __builtin_shufflevector(p0, p1, 0,1,2,3,4,5,6,7,8,9,10,11,12,13,14,15);
    v8h q0 = *reinterpret_cast<const v8h*>(arow1 + kb);
    v8h q1 = *reinterpret_cast<const v8h*>(arow1 + kb + 16);
    v16h a1 = __builtin_shufflevector(q0, q1, 0,1,2,3,4,5,6,7,8,9,10,11,12,13,14,15);
    __builtin_prefetch(arow0 + kb + 32, 0, 3);   // global_prefetch_b8, near cache
    __builtin_prefetch(arow1 + kb + 32, 0, 3);

    // B fragment: lane = N column, K in table order (same as A); reused by 2 WMMAs
    const size_t brow = (size_t)(nbase + (lane & 15)) * DIM + kb + ahalf;
#pragma unroll
    for (int j = 0; j < 4; ++j) {
      v16h bq = __builtin_shufflevector(*reinterpret_cast<const v8h*>(Wqt + brow + (size_t)j * 16 * DIM), *reinterpret_cast<const v8h*>(Wqt + brow + (size_t)j * 16 * DIM + 16), 0,1,2,3,4,5,6,7,8,9,10,11,12,13,14,15);
      cq[0][j] = __builtin_amdgcn_wmma_f32_16x16x32_f16(false, a0, false, bq,
                                                        (short)0, cq[0][j], false, false);
      cq[1][j] = __builtin_amdgcn_wmma_f32_16x16x32_f16(false, a1, false, bq,
                                                        (short)0, cq[1][j], false, false);
      v16h bk = __builtin_shufflevector(*reinterpret_cast<const v8h*>(Wkt + brow + (size_t)j * 16 * DIM), *reinterpret_cast<const v8h*>(Wkt + brow + (size_t)j * 16 * DIM + 16), 0,1,2,3,4,5,6,7,8,9,10,11,12,13,14,15);
      ck[0][j] = __builtin_amdgcn_wmma_f32_16x16x32_f16(false, a0, false, bk,
                                                        (short)0, ck[0][j], false, false);
      ck[1][j] = __builtin_amdgcn_wmma_f32_16x16x32_f16(false, a1, false, bk,
                                                        (short)0, ck[1][j], false, false);
      v16h bv = __builtin_shufflevector(*reinterpret_cast<const v8h*>(Wvt + brow + (size_t)j * 16 * DIM), *reinterpret_cast<const v8h*>(Wvt + brow + (size_t)j * 16 * DIM + 16), 0,1,2,3,4,5,6,7,8,9,10,11,12,13,14,15);
      cv[0][j] = __builtin_amdgcn_wmma_f32_16x16x32_f16(false, a0, false, bv,
                                                        (short)0, cv[0][j], false, false);
      cv[1][j] = __builtin_amdgcn_wmma_f32_16x16x32_f16(false, a1, false, bv,
                                                        (short)0, cv[1][j], false, false);
    }
  }

  // C/D layout: element r, lanes 0-15 -> M = r, N = lane; lanes 16-31 -> M = 8+r
#pragma unroll
  for (int s = 0; s < 2; ++s) {
#pragma unroll
    for (int r = 0; r < 8; ++r) {
      float d = 0.f;
#pragma unroll
      for (int j = 0; j < 4; ++j) d += cq[s][j][r] * ck[s][j][r];
      // reduce over the 16 N-lanes (stays within each half of the wave)
      d += __shfl_xor(d, 1, 32);
      d += __shfl_xor(d, 2, 32);
      d += __shfl_xor(d, 4, 32);
      d += __shfl_xor(d, 8, 32);
      const int row = mtile + s * 16 + r + ((lane >> 4) << 3);
      if ((lane & 15) == 0)
        dots[(size_t)row * HEADS + head] = d * 0.125f;   // 1/sqrt(64)
#pragma unroll
      for (int j = 0; j < 4; ++j)
        vh[(size_t)row * DIM + nbase + j * 16 + (lane & 15)] = (_Float16)cv[s][j][r];
    }
  }
}

// ---------------------------------------------------------- softmax over t=4096

__global__ __launch_bounds__(256)
void softmax_kernel(float* __restrict__ dots) {
  const int b = blockIdx.x >> 4;
  const int h = blockIdx.x & 15;
  const int tid = threadIdx.x;
  __shared__ float red[256];

  float vals[SEQ / 256];
  float m = -1e30f;
#pragma unroll
  for (int i = 0; i < SEQ / 256; ++i) {
    vals[i] = dots[(size_t)(b * SEQ + tid + i * 256) * HEADS + h];
    m = fmaxf(m, vals[i]);
  }
  red[tid] = m; __syncthreads();
  for (int s = 128; s > 0; s >>= 1) {
    if (tid < s) red[tid] = fmaxf(red[tid], red[tid + s]);
    __syncthreads();
  }
  m = red[0]; __syncthreads();

  float sum = 0.f;
#pragma unroll
  for (int i = 0; i < SEQ / 256; ++i) { vals[i] = __expf(vals[i] - m); sum += vals[i]; }
  red[tid] = sum; __syncthreads();
  for (int s = 128; s > 0; s >>= 1) {
    if (tid < s) red[tid] += red[tid + s];
    __syncthreads();
  }
  const float inv = 1.0f / red[0];
#pragma unroll
  for (int i = 0; i < SEQ / 256; ++i)
    dots[(size_t)(b * SEQ + tid + i * 256) * HEADS + h] = vals[i] * inv;
}

// ------------------------- output GEMM: (attn * v) @ Wo^T + bo, permuted store
// A-fragment scaled on the fly: K chunks of 32 never cross the 64-wide head.
// Two M-subtiles per wave; each Wo fragment feeds 2 WMMAs.

__global__ __launch_bounds__(128)
void out_kernel(const _Float16* __restrict__ vh,
                const float* __restrict__ attn,
                const _Float16* __restrict__ Wot,
                const float* __restrict__ bo,
                float* __restrict__ out) {
  const int lane  = threadIdx.x & 31;
  const int wave  = threadIdx.x >> 5;
  const int mtile = (blockIdx.x * 4 + wave) * 32;
  const int ncb   = blockIdx.y * 64;
  const int mrow  = mtile + (lane & 15);
  const int ahalf = (lane < 16) ? 0 : 8;

  v8f zero = {};
  v8f c[2][4];
#pragma unroll
  for (int s = 0; s < 2; ++s)
#pragma unroll
    for (int j = 0; j < 4; ++j) c[s][j] = zero;

  const _Float16* arow0 = vh + (size_t)mrow * DIM + ahalf;
  const _Float16* arow1 = arow0 + (size_t)16 * DIM;
  const float* att0 = attn + (size_t)mrow * HEADS;
  const float* att1 = att0 + (size_t)16 * HEADS;

  for (int kb = 0; kb < DIM; kb += 32) {
    const int h = kb >> 6;                       // head owning this K chunk
    const _Float16 f0 = (_Float16)att0[h];
    const _Float16 f1 = (_Float16)att1[h];
    v8h p0 = *reinterpret_cast<const v8h*>(arow0 + kb) * f0;
    v8h p1 = *reinterpret_cast<const v8h*>(arow0 + kb + 16) * f0;
    v16h a0 = __builtin_shufflevector(p0, p1, 0,1,2,3,4,5,6,7,8,9,10,11,12,13,14,15);
    v8h q0 = *reinterpret_cast<const v8h*>(arow1 + kb) * f1;
    v8h q1 = *reinterpret_cast<const v8h*>(arow1 + kb + 16) * f1;
    v16h a1 = __builtin_shufflevector(q0, q1, 0,1,2,3,4,5,6,7,8,9,10,11,12,13,14,15);

    const size_t brow = (size_t)(ncb + (lane & 15)) * DIM + kb + ahalf;
#pragma unroll
    for (int j = 0; j < 4; ++j) {
      v16h bw = __builtin_shufflevector(*reinterpret_cast<const v8h*>(Wot + brow + (size_t)j * 16 * DIM), *reinterpret_cast<const v8h*>(Wot + brow + (size_t)j * 16 * DIM + 16), 0,1,2,3,4,5,6,7,8,9,10,11,12,13,14,15);
      c[0][j] = __builtin_amdgcn_wmma_f32_16x16x32_f16(false, a0, false, bw,
                                                       (short)0, c[0][j], false, false);
      c[1][j] = __builtin_amdgcn_wmma_f32_16x16x32_f16(false, a1, false, bw,
                                                       (short)0, c[1][j], false, false);
    }
  }

#pragma unroll
  for (int j = 0; j < 4; ++j) {
    const int col = ncb + j * 16 + (lane & 15);
    const float bias = bo[col];
#pragma unroll
    for (int s = 0; s < 2; ++s) {
#pragma unroll
      for (int r = 0; r < 8; ++r) {
        const int m = mtile + s * 16 + r + ((lane >> 4) << 3);
        const int t = m & (SEQ - 1);
        const int bb = m >> 12;
        // strided_transpose: t = i*32 + j  ->  P(t) = (t%32)*128 + t/32
        const int orow = (bb << 12) + ((t & 31) << 7) + (t >> 5);
        out[(size_t)orow * DIM + col] = c[s][j][r] + bias;
      }
    }
  }
}

// -------------------------------------------------------------------- launcher

extern "C" void kernel_launch(void* const* d_in, const int* in_sizes, int n_in,
                              void* d_out, int out_size, void* d_ws, size_t ws_size,
                              hipStream_t stream) {
  const float* x  = (const float*)d_in[0];
  const float* Wq = (const float*)d_in[1];
  const float* Wk = (const float*)d_in[2];
  const float* Wv = (const float*)d_in[3];
  const float* Wo = (const float*)d_in[4];
  const float* bo = (const float*)d_in[5];
  float* out = (float*)d_out;

  char* ws = (char*)d_ws;
  const size_t MB = (size_t)1 << 20;
  _Float16* xh   = (_Float16*)(ws);             // 32 MB  x in f16
  _Float16* Wqt  = (_Float16*)(ws + 32 * MB);   //  2 MB  Wq^T f16
  _Float16* Wkt  = (_Float16*)(ws + 34 * MB);   //  2 MB
  _Float16* Wvt  = (_Float16*)(ws + 36 * MB);   //  2 MB
  _Float16* Wot  = (_Float16*)(ws + 38 * MB);   //  2 MB
  _Float16* vh   = (_Float16*)(ws + 40 * MB);   // 32 MB  v in f16
  float*    dots = (float*)   (ws + 72 * MB);   //  1 MB  dots -> attn (in place)

  cvt_f32_to_f16<<<2048, 256, 0, stream>>>(x, xh, ROWS * DIM);
  transpose_to_f16<<<1024, 256, 0, stream>>>(Wq, Wqt);
  transpose_to_f16<<<1024, 256, 0, stream>>>(Wk, Wkt);
  transpose_to_f16<<<1024, 256, 0, stream>>>(Wv, Wvt);
  transpose_to_f16<<<1024, 256, 0, stream>>>(Wo, Wot);

  dim3 gq(ROWS / 128, HEADS);       // 4 waves/block * 32 rows = 128 rows/block
  qkv_dots_kernel<<<gq, 128, 0, stream>>>(xh, Wqt, Wkt, Wvt, vh, dots);

  softmax_kernel<<<BATCH * HEADS, 256, 0, stream>>>(dots);

  dim3 go(ROWS / 128, DIM / 64);
  out_kernel<<<go, 128, 0, stream>>>(vh, dots, Wot, bo, out);
}
